// ConvolutionalAttention_6700148982325
// MI455X (gfx1250) — matrix operand order repaired and checked by reading, not compiled, not run on hardware
//
#include <hip/hip_runtime.h>

// Problem constants (from reference)
#define NB    8      // batch
#define SEQ   256    // S
#define DIMT  64     // T
#define DCH   100    // D channels
#define DPAD  112    // channels padded to 7*16
#define KPAD  224    // 2*DPAD, 7 k-steps of 32
#define ROWS1 255    // conv1/conv2 output height (S-H+1)
#define ROWS3 254    // conv3 output height
#define TEMPR (1.0f/10.0f)
#define EPS   1e-5f

// Toggle for the CDNA5 async global->LDS staging path (inline asm).
#define USE_ASYNC_LDS 1

typedef __attribute__((ext_vector_type(16))) __bf16 v16bf;
typedef __attribute__((ext_vector_type(8)))  float  v8f;

union BF16Frag { v16bf v; __bf16 h[16]; };

// ---- workspace layout (float offsets unless noted) ----
// Qr   f32 [B][255][112]                       off 0        size 228480
// Kr   f32 [B][256][112]                       off 228480   size 229376
// Wp   bf16[7kt][7mt][32lane][16e]             byte off 1831424 (50176 B)
// cons f32 [7][112] (a1,b1f,b2,a2,b2f,w3a,w3b) off 470400   size 784
// cA   f32 [B][255][256]                       off 471184   size 522240
// cB   f32 [B][255][256]                       off 993424   size 522240
#define QR_OFF   0
#define KR_OFF   228480
#define WP_BYTE  1831424
#define CONS_OFF 470400
#define CA_OFF   471184
#define CB_OFF   993424

// ---------------------------------------------------------------------------
// Fused BN / conv3 constants, padded to 112 channels (pad lanes -> exact 0)
__global__ void ca_prep_consts(const float* g1, const float* be1, const float* m1,
                               const float* v1, const float* b2, const float* g2,
                               const float* be2, const float* m2, const float* v2,
                               const float* W3, float* cons) {
  int d = threadIdx.x;
  if (d >= DPAD) return;
  bool ok = d < DCH;
  float a1 = ok ? g1[d] * rsqrtf(v1[d] + EPS) : 0.0f;
  float b1f = ok ? be1[d] - m1[d] * a1 : 0.0f;
  float a2 = ok ? g2[d] * rsqrtf(v2[d] + EPS) : 0.0f;
  float b2f = ok ? be2[d] - m2[d] * a2 : 0.0f;
  cons[0 * DPAD + d] = a1;                       // alpha1
  cons[1 * DPAD + d] = b1f;                      // beta1
  cons[2 * DPAD + d] = ok ? b2[d] : 0.0f;        // conv2 bias
  cons[3 * DPAD + d] = a2;                       // alpha2
  cons[4 * DPAD + d] = b2f;                      // beta2
  cons[5 * DPAD + d] = ok ? W3[d * 2 + 0] : 0.0f;
  cons[6 * DPAD + d] = ok ? W3[d * 2 + 1] : 0.0f;
}

// ---------------------------------------------------------------------------
// Pack W2 into the exact A-fragment layout (16-bit A 16x32, ISA 7.12.2):
// lane l: m = l&15, K = (e&7) + 8*(l>>4) + 16*(e>>3) within a 32-K step.
// Logical K index: kk = h*112 + c.
__global__ void ca_pack_w2(const float* W2, __bf16* wp) {
  int idx = blockIdx.x * blockDim.x + threadIdx.x;
  if (idx >= 7 * 7 * 32 * 16) return;
  int e    = idx & 15;
  int lin  = idx >> 4;
  int lane = lin & 31;
  int fm   = lin >> 5;
  int mt   = fm % 7;
  int kt   = fm / 7;
  int K  = kt * 32 + (e & 7) + 8 * (lane >> 4) + 16 * (e >> 3);
  int d  = mt * 16 + (lane & 15);
  int h  = K / DPAD;
  int c  = K - h * DPAD;
  float v = (c < DCH && d < DCH) ? W2[(d * DCH + c) * 2 + h] : 0.0f;
  wp[idx] = (__bf16)v;
}

// ---------------------------------------------------------------------------
// Separable conv1 factors.
__global__ void ca_prep_q(const float* q, const float* W1, float* Qr) {
  int d = threadIdx.x, i = blockIdx.x, b = blockIdx.y;
  if (d >= DPAD) return;
  float acc = 0.0f;
  if (d < DCH) {
    const float* q0 = q + (b * SEQ + i) * DIMT;
    for (int t = 0; t < DIMT; ++t)
      acc += q0[t] * W1[(d * 128 + t) * 2 + 0] + q0[DIMT + t] * W1[(d * 128 + t) * 2 + 1];
  }
  Qr[(b * ROWS1 + i) * DPAD + d] = acc;
}

__global__ void ca_prep_k(const float* k, const float* W1, const float* b1, float* Kr) {
  int d = threadIdx.x, j = blockIdx.x, b = blockIdx.y;
  if (d >= DPAD) return;
  float acc = 0.0f;
  if (d < DCH) {
    acc = b1[d];
    const float* k0 = k + (b * SEQ + j) * DIMT;
    for (int t = 0; t < DIMT; ++t)
      acc += k0[t] * (W1[(d * 128 + 64 + t) * 2 + 0] + W1[(d * 128 + 64 + t) * 2 + 1]);
  }
  Kr[(b * SEQ + j) * DPAD + d] = acc;
}

// ---------------------------------------------------------------------------
// Fused conv2 (bf16 WMMA) + bias + residual + BN2 + ReLU + conv3 partial dots.
// Workgroup: (jblock, i, b); 4 waves, each owns one 16-wide N tile, 7 M tiles.
__global__ __launch_bounds__(128)
void ca_conv2_fused(const float* __restrict__ Qr, const float* __restrict__ Kr,
                    const __bf16* __restrict__ wp, const float* __restrict__ cons,
                    float* __restrict__ cA, float* __restrict__ cB) {
  const int b = blockIdx.z, i = blockIdx.y, j0 = blockIdx.x * 64;
  const int tid = threadIdx.x;

  __shared__ float kls[64 * 113];      // raw Kr slice, padded stride vs banks
  __shared__ float qvs[2 * DPAD];      // alpha1*Q1[p-1] + beta1 per h
  __shared__ float kqs[2 * DPAD];      // alpha1 (0 for the padded top row)
  __shared__ float qraw[DPAD];         // Q1 row i (residual)
  __shared__ float cs[7 * DPAD];       // fused constants

#if USE_ASYNC_LDS
  // CDNA5 async copy: cache -> LDS directly (no VGPR round trip), ASYNCcnt.
  // 64 rows x 112 floats; 28 x b128 chunks per row; LDS row stride 113 floats.
  {
    const char* gbase = (const char*)(Kr + (size_t)(b * SEQ + j0) * DPAD);
    for (int idx = tid; idx < 64 * 28; idx += 128) {
      int row = idx / 28, cc = idx - row * 28;
      unsigned long long g64 = (unsigned long long)(gbase + row * 448 + cc * 16);
      unsigned lds32 = (unsigned)(unsigned long long)&kls[row * 113 + cc * 4];
      asm volatile("global_load_async_to_lds_b128 %0, %1, off"
                   :: "v"(lds32), "v"(g64) : "memory");
    }
  }
#else
  for (int idx = tid; idx < 64 * DPAD; idx += 128) {
    int j = idx / DPAD, c = idx - j * DPAD;
    kls[j * 113 + c] = Kr[(b * SEQ + j0) * DPAD + idx];
  }
#endif

  for (int idx = tid; idx < 7 * DPAD; idx += 128) cs[idx] = cons[idx];
  if (tid < DPAD) {
    int c = tid;
    float a1 = cons[0 * DPAD + c], b1f = cons[1 * DPAD + c];
    float qm1 = (i == 0) ? 0.0f : Qr[(b * ROWS1 + (i - 1)) * DPAD + c];
    qvs[c]        = (i == 0) ? b1f : a1 * qm1 + b1f;   // h=0 row (p=i)
    kqs[c]        = (i == 0) ? 0.0f : a1;
    float q0 = Qr[(b * ROWS1 + i) * DPAD + c];
    qvs[DPAD + c] = a1 * q0 + b1f;                     // h=1 row (p=i+1)
    kqs[DPAD + c] = a1;
    qraw[c] = q0;
  }
#if USE_ASYNC_LDS
  asm volatile("s_wait_asynccnt 0" ::: "memory");      // own transfers done
#endif
  __syncthreads();                                     // everyone's done

  const int lane = tid & 31, g = lane >> 4, n = lane & 15;
  const int jl = (tid >> 5) * 16 + n;      // local j this lane accumulates for

  __builtin_prefetch(wp, 0, 3);            // global_prefetch_b8 on hot weights

  v8f acc[7];
#pragma unroll
  for (int mt = 0; mt < 7; ++mt) acc[mt] = (v8f){0,0,0,0,0,0,0,0};

#pragma unroll
  for (int kt = 0; kt < 7; ++kt) {
    // Build B fragment (K x 16) on the fly: relu1[c, i+h, j] in bf16.
    BF16Frag bu;
#pragma unroll
    for (int e = 0; e < 16; ++e) {
      int kk = kt * 32 + (e & 7) + 8 * g + 16 * (e >> 3);   // same K order as packed A
      int h  = kk >= DPAD;
      int c  = kk - h * DPAD;
      float v = qvs[h * DPAD + c] + kqs[h * DPAD + c] * kls[jl * 113 + c];
      bu.h[e] = (__bf16)fmaxf(v, 0.0f);
    }
#pragma unroll
    for (int mt = 0; mt < 7; ++mt) {
      v16bf af = *(const v16bf*)(wp + (size_t)(((kt * 7 + mt) * 32 + lane) * 16));
      acc[mt] = __builtin_amdgcn_wmma_f32_16x16x32_bf16(
          false, af, false, bu.v, (short)0, acc[mt], false, false);
    }
  }

  // Epilogue: +b2, +residual (Q1+K1+b1), BN2, ReLU, conv3 partial dots over d.
  float sA = 0.0f, sB = 0.0f;
#pragma unroll
  for (int mt = 0; mt < 7; ++mt) {
#pragma unroll
    for (int r = 0; r < 8; ++r) {
      int d = mt * 16 + r + 8 * g;
      float a2v  = acc[mt][r] + cs[2 * DPAD + d];
      float resv = qraw[d] + kls[jl * 113 + d];
      float x    = cs[3 * DPAD + d] * (a2v + resv) + cs[4 * DPAD + d];
      float r2   = fmaxf(x, 0.0f);
      sA += r2 * cs[5 * DPAD + d];
      sB += r2 * cs[6 * DPAD + d];
    }
  }
  sA += __shfl_xor(sA, 16);                // fold the two d half-lanes (wave32)
  sB += __shfl_xor(sB, 16);
  if (g == 0) {
    int o = (b * ROWS1 + i) * SEQ + j0 + jl;
    cA[o] = sA;
    cB[o] = sB;
  }
}

// ---------------------------------------------------------------------------
// a3 = cA[r] + cB[r+1] + b3, /TEMP, softmax over j -> attn
__global__ __launch_bounds__(256)
void ca_softmax(const float* __restrict__ cA, const float* __restrict__ cB,
                const float* __restrict__ b3, float* __restrict__ attn) {
  const int b = blockIdx.y, r = blockIdx.x, j = threadIdx.x;
  __shared__ float red[256];
  float logit = (cA[(b * ROWS1 + r) * SEQ + j] + cB[(b * ROWS1 + r + 1) * SEQ + j] + b3[0]) * TEMPR;
  red[j] = logit;
  __syncthreads();
  for (int s = 128; s > 0; s >>= 1) { if (j < s) red[j] = fmaxf(red[j], red[j + s]); __syncthreads(); }
  float m = red[0];
  __syncthreads();
  float ev = __expf(logit - m);
  red[j] = ev;
  __syncthreads();
  for (int s = 128; s > 0; s >>= 1) { if (j < s) red[j] += red[j + s]; __syncthreads(); }
  attn[(b * ROWS3 + r) * SEQ + j] = ev / red[0];
}

// ---------------------------------------------------------------------------
// out = attn @ v via bf16 WMMA. Grid (16 row-tiles, B); 4 waves -> 4 t-tiles.
__global__ __launch_bounds__(128)
void ca_attn_v(const float* __restrict__ attn, const float* __restrict__ v,
               float* __restrict__ out) {
  const int b = blockIdx.y, rt = blockIdx.x;
  const int lane = threadIdx.x & 31, g = lane >> 4, n = lane & 15;
  const int wt = threadIdx.x >> 5;         // t tile

  v8f acc = (v8f){0,0,0,0,0,0,0,0};
#pragma unroll
  for (int kt = 0; kt < 8; ++kt) {
    BF16Frag au, bu;
    int r = rt * 16 + n;
    bool valid = r < ROWS3;
    const float* arow = attn + (size_t)(b * ROWS3 + (valid ? r : 0)) * SEQ;
#pragma unroll
    for (int e = 0; e < 16; ++e) {
      int jA = kt * 32 + (e & 7) + 8 * g + 16 * (e >> 3);   // A-layout K index
      au.h[e] = (__bf16)(valid ? arow[jA] : 0.0f);
      int jB = kt * 32 + (e & 7) + 8 * g + 16 * (e >> 3);   // same K order as A
      bu.h[e] = (__bf16)v[(b * SEQ + jB) * DIMT + wt * 16 + n];
    }
    acc = __builtin_amdgcn_wmma_f32_16x16x32_bf16(
        false, au.v, false, bu.v, (short)0, acc, false, false);
  }
#pragma unroll
  for (int r = 0; r < 8; ++r) {
    int row = rt * 16 + r + 8 * g;
    if (row < ROWS3) out[(b * ROWS3 + row) * DIMT + wt * 16 + n] = acc[r];
  }
}

// ---------------------------------------------------------------------------
extern "C" void kernel_launch(void* const* d_in, const int* in_sizes, int n_in,
                              void* d_out, int out_size, void* d_ws, size_t ws_size,
                              hipStream_t stream) {
  const float* q   = (const float*)d_in[0];
  const float* k   = (const float*)d_in[1];
  const float* vin = (const float*)d_in[2];
  const float* W1  = (const float*)d_in[3];
  const float* b1  = (const float*)d_in[4];
  const float* g1  = (const float*)d_in[5];
  const float* be1 = (const float*)d_in[6];
  const float* m1  = (const float*)d_in[7];
  const float* v1  = (const float*)d_in[8];
  const float* W2  = (const float*)d_in[9];
  const float* b2  = (const float*)d_in[10];
  const float* g2  = (const float*)d_in[11];
  const float* be2 = (const float*)d_in[12];
  const float* m2  = (const float*)d_in[13];
  const float* v2  = (const float*)d_in[14];
  const float* W3  = (const float*)d_in[15];
  const float* b3  = (const float*)d_in[16];

  float*  ws   = (float*)d_ws;
  float*  Qr   = ws + QR_OFF;
  float*  Kr   = ws + KR_OFF;
  __bf16* wp   = (__bf16*)((char*)d_ws + WP_BYTE);
  float*  cons = ws + CONS_OFF;
  float*  cA   = ws + CA_OFF;
  float*  cB   = ws + CB_OFF;

  float* out  = (float*)d_out;                 // (B,254,64)
  float* attn = out + NB * ROWS3 * DIMT;       // (B,254,256)

  ca_prep_consts<<<1, 128, 0, stream>>>(g1, be1, m1, v1, b2, g2, be2, m2, v2, W3, cons);
  ca_pack_w2<<<(7 * 7 * 32 * 16 + 255) / 256, 256, 0, stream>>>(W2, wp);
  ca_prep_q<<<dim3(ROWS1, NB), DPAD, 0, stream>>>(q, W1, Qr);
  ca_prep_k<<<dim3(SEQ, NB), DPAD, 0, stream>>>(k, W1, b1, Kr);
  ca_conv2_fused<<<dim3(4, ROWS1, NB), 128, 0, stream>>>(Qr, Kr, wp, cons, cA, cB);
  ca_softmax<<<dim3(ROWS3, NB), 256, 0, stream>>>(cA, cB, b3, attn);
  ca_attn_v<<<dim3(16, NB), 128, 0, stream>>>(attn, vin, out);
}
